// TransformerBlock_8804682957057
// MI455X (gfx1250) — compile-verified, matrix operand order checked by reading, not run on hardware
//
#include <hip/hip_runtime.h>
#include <hip/hip_bf16.h>
#include <cmath>

// ---------------------------------------------------------------------------
// Types for CDNA5 WMMA (gfx1250, wave32)
// ---------------------------------------------------------------------------
typedef __attribute__((ext_vector_type(16))) __bf16 v16bf;
typedef __attribute__((ext_vector_type(8)))  __bf16 v8bf;
typedef __attribute__((ext_vector_type(8)))  float  v8f;

typedef __attribute__((address_space(3))) void lds_void;

#define DEV static __device__ __forceinline__

DEV __bf16 f2bf(float f) {
    union { float f; unsigned u; } a; a.f = f;
    unsigned r = a.u + 0x7FFFu + ((a.u >> 16) & 1u);
    unsigned short h = (unsigned short)(r >> 16);
    union { unsigned short s; __bf16 b; } b2; b2.s = h;
    return b2.b;
}

DEV v8f wmma_bf16(const v16bf& a, const v16bf& b, const v8f& c) {
    return __builtin_amdgcn_wmma_f32_16x16x32_bf16(
        /*neg_a=*/false, a, /*neg_b=*/false, b,
        /*c_mod=*/(short)0, c, /*reuse_a=*/false, /*reuse_b=*/false);
}

// Async global -> LDS copy of 16 bytes (GVS mode: sgpr base + vgpr byte off).
// Tracked by ASYNCcnt; caller must s_wait_asynccnt before consuming LDS.
DEV void async_b128(void* lds_dst, const void* gbase, unsigned byte_off) {
    asm volatile("global_load_async_to_lds_b128 %0, %1, %2"
                 :
                 : "v"((lds_void*)lds_dst), "v"(byte_off), "s"(gbase)
                 : "memory");
}

DEV void wait_async0() {
    asm volatile("s_wait_asynccnt 0x0" ::: "memory");
}

// Build a 16-element bf16 fragment from two 8-half LDS chunks
// (VGPR k holds elements 2k,2k+1: elems [0..7] = K base..base+7,
//  elems [8..15] = K base+16..base+23 per CDNA5 ISA layout)
DEV v16bf frag_from(const __bf16* lo, const __bf16* hi) {
    v8bf l = *(const v8bf*)lo;
    v8bf h = *(const v8bf*)hi;
    v16bf r;
#pragma unroll
    for (int i = 0; i < 8; ++i) { r[i] = l[i]; r[i + 8] = h[i]; }
    return r;
}

// ---------------------------------------------------------------------------
// Problem constants
// ---------------------------------------------------------------------------
static constexpr int Bc = 2, Sc = 2048, Dc = 768, Hc = 12, FFc = 3072;
static constexpr int HDc = 64;
static constexpr int Mc = Bc * Sc;   // 4096 flattened rows

// ---------------------------------------------------------------------------
// f32 -> bf16 conversion
// ---------------------------------------------------------------------------
__global__ void k_cvt_bf16(const float* __restrict__ src,
                           __bf16* __restrict__ dst, int n) {
    int i = blockIdx.x * (blockDim.x * 4) + threadIdx.x;
#pragma unroll
    for (int t = 0; t < 4; ++t) {
        int j = i + t * blockDim.x;
        if (j < n) dst[j] = f2bf(src[j]);
    }
}

// ---------------------------------------------------------------------------
// Tiled WMMA GEMM:  C(MxN) = A(MxK) * W(NxK)^T + bias  [opt relu]
// WG: 256 thr = 8 waves; tile 128x64; BK=32; wave -> 32x32 (2x2 frags)
// Tiles staged into LDS with GLOBAL_LOAD_ASYNC_TO_LDS_B128 (ASYNCcnt).
// M%128==0, N%64==0, K%32==0 guaranteed by the shapes used here.
// ---------------------------------------------------------------------------
template <bool OUT_BF16, bool RELU>
__global__ __launch_bounds__(256)
void k_gemm(const __bf16* __restrict__ A, const __bf16* __restrict__ W,
            const float* __restrict__ bias,
            float* __restrict__ Cf, __bf16* __restrict__ Ch,
            int M, int N, int K) {
    constexpr int BM = 128, BN = 64, BK = 32, PAD = 8;
    __shared__ alignas(16) __bf16 As[BM][BK + PAD];
    __shared__ alignas(16) __bf16 Bs[BN][BK + PAD];

    const int tid  = threadIdx.x;
    const int wave = tid >> 5;
    const int lane = tid & 31;
    const int l16  = lane & 15;
    const int lhi  = lane >> 4;          // 0 | 1 : which K-half this lane holds
    const int wr   = wave >> 1;          // 0..3
    const int wc   = wave & 1;           // 0..1
    const int m_wg = blockIdx.y * BM;
    const int n_wg = blockIdx.x * BN;

    v8f acc[2][2];
#pragma unroll
    for (int i = 0; i < 2; ++i)
#pragma unroll
        for (int j = 0; j < 2; ++j) acc[i][j] = (v8f)0.0f;

    for (int k0 = 0; k0 < K; k0 += BK) {
        // ---- A tile: 128x32 halfs = 512 async chunks of 16B ----
#pragma unroll
        for (int c = 0; c < 2; ++c) {
            int chunk = tid * 2 + c;
            int r  = chunk >> 2;
            int cc = (chunk & 3) * 8;
            unsigned go = ((unsigned)(m_wg + r) * (unsigned)K +
                           (unsigned)(k0 + cc)) * 2u;
            async_b128(&As[r][cc], A, go);
        }
        // ---- B tile: 64x32 halfs = 256 async chunks ----
        {
            int r  = tid >> 2;
            int cc = (tid & 3) * 8;
            unsigned go = ((unsigned)(n_wg + r) * (unsigned)K +
                           (unsigned)(k0 + cc)) * 2u;
            async_b128(&Bs[r][cc], W, go);
        }
        // prefetch next k-tile (global_prefetch_b8)
        if (k0 + BK < K) {
            __builtin_prefetch(A + (size_t)(m_wg + (tid >> 1)) * K + k0 + BK, 0, 1);
            __builtin_prefetch(W + (size_t)(n_wg + (tid >> 2)) * K + k0 + BK, 0, 1);
        }
        wait_async0();
        __syncthreads();

        const int kb = lhi * 8;          // per-lane K base within tile
        v16bf af[2], bf[2];
#pragma unroll
        for (int fr = 0; fr < 2; ++fr) {
            int row = wr * 32 + fr * 16 + l16;
            af[fr] = frag_from(&As[row][kb], &As[row][kb + 16]);
        }
#pragma unroll
        for (int fc = 0; fc < 2; ++fc) {
            int row = wc * 32 + fc * 16 + l16;
            bf[fc] = frag_from(&Bs[row][kb], &Bs[row][kb + 16]);
        }
#pragma unroll
        for (int fr = 0; fr < 2; ++fr)
#pragma unroll
            for (int fc = 0; fc < 2; ++fc)
                acc[fr][fc] = wmma_bf16(af[fr], bf[fc], acc[fr][fc]);
        __syncthreads();
    }

    // ---- epilogue: C-frag element j -> row (lhi*8+j), col l16 ----
#pragma unroll
    for (int fr = 0; fr < 2; ++fr)
#pragma unroll
        for (int fc = 0; fc < 2; ++fc) {
            int cidx = n_wg + wc * 32 + fc * 16 + l16;
            float bv = bias[cidx];
#pragma unroll
            for (int j = 0; j < 8; ++j) {
                int r = m_wg + wr * 32 + fr * 16 + lhi * 8 + j;
                float v = acc[fr][fc][j] + bv;
                if (RELU) v = fmaxf(v, 0.0f);
                if (OUT_BF16) Ch[(size_t)r * N + cidx] = f2bf(v);
                else          Cf[(size_t)r * N + cidx] = v;
            }
        }
}

// ---------------------------------------------------------------------------
// Flash attention with ALiBi + causal mask (recomputed analytically).
// Grid: (B*H, S/128). 256 thr = 8 waves; wave w owns q rows [q0+16w, q0+16w+16).
// k-tile = 32 keys. scores: 4 WMMA / tile / wave; PV: 4 WMMA / tile / wave.
// K tile via async-to-LDS; V tile transposed through VGPRs.
// ---------------------------------------------------------------------------
__global__ __launch_bounds__(256)
void k_attn(const __bf16* __restrict__ Qh, const __bf16* __restrict__ Kh,
            const __bf16* __restrict__ Vh, __bf16* __restrict__ Ctx) {
    const int bh = blockIdx.x;
    const int b  = bh / Hc;
    const int h  = bh % Hc;
    const int q0 = blockIdx.y * 128;

    const int tid  = threadIdx.x;
    const int wave = tid >> 5;
    const int lane = tid & 31;
    const int l16  = lane & 15;
    const int lhi  = lane >> 4;
    const float slope = exp2f(-8.0f * (float)(h + 1) / (float)Hc);

    __shared__ alignas(16) __bf16 Ks[32][HDc + 8];       // key x d
    __shared__ alignas(16) __bf16 Vt[HDc][32 + 8];       // d x key (transposed)
    __shared__ alignas(16) __bf16 Ps[8][16][32 + 8];     // per-wave P staging

    // ---- load this lane's Q A-fragments (16 x 64 -> 2 ksteps) ----
    const int qrow = q0 + wave * 16 + l16;
    v16bf qa[2];
#pragma unroll
    for (int ks = 0; ks < 2; ++ks) {
        const __bf16* qp = Qh + ((size_t)(b * Sc + qrow)) * Dc + h * HDc
                         + ks * 32 + lhi * 8;
        qa[ks] = frag_from(qp, qp + 16);
    }

    float mrow[8], lrow[8];
    v8f of[4];
#pragma unroll
    for (int j = 0; j < 8; ++j) { mrow[j] = -INFINITY; lrow[j] = 0.0f; }
#pragma unroll
    for (int f = 0; f < 4; ++f) of[f] = (v8f)0.0f;

    const int ntiles = q0 / 32 + 4;                      // causal bound
    for (int jt = 0; jt < ntiles; ++jt) {
        const int kbase = jt * 32;
        // ---- K tile: 32 keys x 64 d, async global -> LDS ----
        {
            int r  = tid >> 3;
            int cc = (tid & 7) * 8;
            unsigned go = ((unsigned)(b * Sc + kbase + r) * (unsigned)Dc +
                           (unsigned)(h * HDc + cc)) * 2u;
            async_b128(&Ks[r][cc], Kh, go);
        }
        // ---- V tile transposed into LDS: Vt[d][key] ----
        {
            int key = tid >> 3;
            int d0  = (tid & 7) * 8;
            v8bf vv = *(const v8bf*)(Vh + ((size_t)(b * Sc + kbase + key)) * Dc
                                     + h * HDc + d0);
#pragma unroll
            for (int i = 0; i < 8; ++i) Vt[d0 + i][key] = vv[i];
        }
        wait_async0();
        __syncthreads();

        // ---- scores S = Q K^T  (2 col-frags x 2 ksteps) ----
        const int kb = lhi * 8;
        v8f sc[2];
#pragma unroll
        for (int fc = 0; fc < 2; ++fc) {
            sc[fc] = (v8f)0.0f;
#pragma unroll
            for (int ks = 0; ks < 2; ++ks) {
                int krow = fc * 16 + l16;
                v16bf bfrag = frag_from(&Ks[krow][ks * 32 + kb],
                                        &Ks[krow][ks * 32 + kb + 16]);
                sc[fc] = wmma_bf16(qa[ks], bfrag, sc[fc]);
            }
        }

        // ---- scale + ALiBi + causal mask ----
#pragma unroll
        for (int fc = 0; fc < 2; ++fc)
#pragma unroll
            for (int j = 0; j < 8; ++j) {
                int qg = q0 + wave * 16 + lhi * 8 + j;
                int kg = kbase + fc * 16 + l16;
                float s = sc[fc][j] * 0.125f + slope * (float)(kg - qg);
                if (kg > qg) s += -1e9f;
                sc[fc][j] = s;
            }

        // ---- online softmax (row reductions across 16-lane halves) ----
#pragma unroll
        for (int j = 0; j < 8; ++j) {
            float mx = fmaxf(sc[0][j], sc[1][j]);
#pragma unroll
            for (int m = 1; m < 16; m <<= 1)
                mx = fmaxf(mx, __shfl_xor(mx, m, 32));
            float mnew  = fmaxf(mrow[j], mx);
            float alpha = __expf(mrow[j] - mnew);
            mrow[j] = mnew;
            float p0 = __expf(sc[0][j] - mnew);
            float p1 = __expf(sc[1][j] - mnew);
            sc[0][j] = p0; sc[1][j] = p1;
            float rs = p0 + p1;
#pragma unroll
            for (int m = 1; m < 16; m <<= 1)
                rs += __shfl_xor(rs, m, 32);
            lrow[j] = lrow[j] * alpha + rs;
#pragma unroll
            for (int f = 0; f < 4; ++f) of[f][j] *= alpha;
        }

        // ---- stage P (C-layout -> LDS) then reload as A-fragment ----
#pragma unroll
        for (int fc = 0; fc < 2; ++fc)
#pragma unroll
            for (int j = 0; j < 8; ++j)
                Ps[wave][lhi * 8 + j][fc * 16 + l16] = f2bf(sc[fc][j]);

        v16bf pa = frag_from(&Ps[wave][l16][kb], &Ps[wave][l16][kb + 16]);

        // ---- O += P V  (contraction over 32 keys, 4 d-col frags) ----
#pragma unroll
        for (int f = 0; f < 4; ++f) {
            int drow = f * 16 + l16;
            v16bf bfrag = frag_from(&Vt[drow][kb], &Vt[drow][kb + 16]);
            of[f] = wmma_bf16(pa, bfrag, of[f]);
        }
        __syncthreads();
    }

    // ---- normalize and write context (B,S,H,HD) as bf16 ----
#pragma unroll
    for (int f = 0; f < 4; ++f)
#pragma unroll
        for (int j = 0; j < 8; ++j) {
            int qg = q0 + wave * 16 + lhi * 8 + j;
            int dc = f * 16 + l16;
            Ctx[((size_t)(b * Sc + qg)) * Dc + h * HDc + dc] =
                f2bf(of[f][j] / lrow[j]);
        }
}

// ---------------------------------------------------------------------------
// Residual + LayerNorm. One row (768) per block of 256 threads.
// Writes f32 output and (optionally) a bf16 copy for the next GEMM.
// ---------------------------------------------------------------------------
__global__ __launch_bounds__(256)
void k_ln_res(const float* __restrict__ X, const float* __restrict__ R,
              const float* __restrict__ g, const float* __restrict__ be,
              float* __restrict__ Yf, __bf16* __restrict__ Yh) {
    const int row  = blockIdx.x;
    const int tid  = threadIdx.x;
    const int wave = tid >> 5;
    const int lane = tid & 31;
    const float* xr = X + (size_t)row * Dc;
    const float* rr = R + (size_t)row * Dc;

    float vals[3], s = 0.0f, s2 = 0.0f;
#pragma unroll
    for (int t = 0; t < 3; ++t) {
        int i = tid + t * 256;
        float v = xr[i] + rr[i];
        vals[t] = v; s += v; s2 += v * v;
    }
#pragma unroll
    for (int m = 1; m < 32; m <<= 1) {
        s  += __shfl_xor(s,  m, 32);
        s2 += __shfl_xor(s2, m, 32);
    }
    __shared__ float ws1[8], ws2[8];
    if (lane == 0) { ws1[wave] = s; ws2[wave] = s2; }
    __syncthreads();
    float ts = 0.0f, ts2 = 0.0f;
#pragma unroll
    for (int w = 0; w < 8; ++w) { ts += ws1[w]; ts2 += ws2[w]; }
    const float mu  = ts * (1.0f / Dc);
    const float var = ts2 * (1.0f / Dc) - mu * mu;
    const float inv = rsqrtf(var + 1e-5f);
#pragma unroll
    for (int t = 0; t < 3; ++t) {
        int i = tid + t * 256;
        float y = (vals[t] - mu) * inv * g[i] + be[i];
        Yf[(size_t)row * Dc + i] = y;
        if (Yh) Yh[(size_t)row * Dc + i] = f2bf(y);
    }
}

// ---------------------------------------------------------------------------
// Host-side orchestration
// ---------------------------------------------------------------------------
static constexpr size_t SZ_BSD_BF  = (size_t)Mc * Dc * 2;     //  6,291,456
static constexpr size_t SZ_WDD_BF  = (size_t)Dc * Dc * 2;     //  1,179,648
static constexpr size_t SZ_WFD_BF  = (size_t)FFc * Dc * 2;    //  4,718,592
static constexpr size_t SZ_BSD_F   = (size_t)Mc * Dc * 4;     // 12,582,912

static constexpr size_t OFF_XH  = 0;                          // reused by FFH
static constexpr size_t OFF_QH  = OFF_XH + SZ_BSD_BF;
static constexpr size_t OFF_KH  = OFF_QH + SZ_BSD_BF;
static constexpr size_t OFF_VH  = OFF_KH + SZ_BSD_BF;
static constexpr size_t OFF_FFH = 0;                          // 25,165,824 B overlay
static constexpr size_t OFF_WQH = OFF_VH + SZ_BSD_BF;         // = 25,165,824
static constexpr size_t OFF_WKH = OFF_WQH + SZ_WDD_BF;
static constexpr size_t OFF_WVH = OFF_WKH + SZ_WDD_BF;
static constexpr size_t OFF_WOH = OFF_WVH + SZ_WDD_BF;
static constexpr size_t OFF_W1H = OFF_WOH + SZ_WDD_BF;
static constexpr size_t OFF_W2H = OFF_W1H + SZ_WFD_BF;
static constexpr size_t OFF_CTX = OFF_W2H + SZ_WFD_BF;
static constexpr size_t OFF_AO  = OFF_CTX + SZ_BSD_BF;        // attn_out, reused as ff2
static constexpr size_t OFF_X1F = OFF_AO  + SZ_BSD_F;
static constexpr size_t OFF_X1H = OFF_X1F + SZ_BSD_F;

extern "C" void kernel_launch(void* const* d_in, const int* in_sizes, int n_in,
                              void* d_out, int out_size, void* d_ws, size_t ws_size,
                              hipStream_t stream) {
    const float* x   = (const float*)d_in[0];
    const float* wq  = (const float*)d_in[3];
    const float* bq  = (const float*)d_in[4];
    const float* wk  = (const float*)d_in[5];
    const float* bk  = (const float*)d_in[6];
    const float* wv  = (const float*)d_in[7];
    const float* bv  = (const float*)d_in[8];
    const float* wo  = (const float*)d_in[9];
    const float* bo  = (const float*)d_in[10];
    const float* w1  = (const float*)d_in[11];
    const float* b1  = (const float*)d_in[12];
    const float* w2  = (const float*)d_in[13];
    const float* b2  = (const float*)d_in[14];
    const float* g1  = (const float*)d_in[15];
    const float* be1 = (const float*)d_in[16];
    const float* g2  = (const float*)d_in[17];
    const float* be2 = (const float*)d_in[18];

    char* ws = (char*)d_ws;
    __bf16* xh  = (__bf16*)(ws + OFF_XH);
    __bf16* qh  = (__bf16*)(ws + OFF_QH);
    __bf16* kh  = (__bf16*)(ws + OFF_KH);
    __bf16* vh  = (__bf16*)(ws + OFF_VH);
    __bf16* ffh = (__bf16*)(ws + OFF_FFH);
    __bf16* wqh = (__bf16*)(ws + OFF_WQH);
    __bf16* wkh = (__bf16*)(ws + OFF_WKH);
    __bf16* wvh = (__bf16*)(ws + OFF_WVH);
    __bf16* woh = (__bf16*)(ws + OFF_WOH);
    __bf16* w1h = (__bf16*)(ws + OFF_W1H);
    __bf16* w2h = (__bf16*)(ws + OFF_W2H);
    __bf16* ctx = (__bf16*)(ws + OFF_CTX);
    float*  ao  = (float*) (ws + OFF_AO);    // attn projection out, then ff2
    float*  x1f = (float*) (ws + OFF_X1F);
    __bf16* x1h = (__bf16*)(ws + OFF_X1H);

    const dim3 blk(256);
    auto cvt = [&](const float* s, __bf16* d, int n) {
        k_cvt_bf16<<<dim3((n + 1023) / 1024), blk, 0, stream>>>(s, d, n);
    };

    // 1) f32 -> bf16 conversions
    cvt(x,  xh,  Mc * Dc);
    cvt(wq, wqh, Dc * Dc);
    cvt(wk, wkh, Dc * Dc);
    cvt(wv, wvh, Dc * Dc);
    cvt(wo, woh, Dc * Dc);
    cvt(w1, w1h, FFc * Dc);
    cvt(w2, w2h, Dc * FFc);

    const dim3 gD(Dc / 64, Mc / 128);     // N=768
    const dim3 gF(FFc / 64, Mc / 128);    // N=3072

    // 2) QKV projections (bf16 out)
    k_gemm<true,  false><<<gD, blk, 0, stream>>>(xh, wqh, bq, nullptr, qh, Mc, Dc, Dc);
    k_gemm<true,  false><<<gD, blk, 0, stream>>>(xh, wkh, bk, nullptr, kh, Mc, Dc, Dc);
    k_gemm<true,  false><<<gD, blk, 0, stream>>>(xh, wvh, bv, nullptr, vh, Mc, Dc, Dc);

    // 3) attention (flash, ALiBi + causal)
    k_attn<<<dim3(Bc * Hc, Sc / 128), blk, 0, stream>>>(qh, kh, vh, ctx);

    // 4) output projection (f32 out)
    k_gemm<false, false><<<gD, blk, 0, stream>>>(ctx, woh, bo, ao, nullptr, Mc, Dc, Dc);

    // 5) x1 = LN(x + attn_out)
    k_ln_res<<<dim3(Mc), blk, 0, stream>>>(x, ao, g1, be1, x1f, x1h);

    // 6) FFN
    k_gemm<true,  true ><<<gF, blk, 0, stream>>>(x1h, w1h, b1, nullptr, ffh, Mc, FFc, Dc);
    k_gemm<false, false><<<gD, blk, 0, stream>>>(ffh, w2h, b2, ao, nullptr, Mc, Dc, FFc);

    // 7) out = LN(x1 + ff)
    k_ln_res<<<dim3(Mc), blk, 0, stream>>>(x1f, ao, g2, be2, (float*)d_out, nullptr);
}
